// SCAM_15367392985855
// MI455X (gfx1250) — matrix operand order repaired and checked by reading, not compiled, not run on hardware
//
#include <hip/hip_runtime.h>
#include <hip/hip_bf16.h>

typedef _Float16 f16;
typedef __attribute__((ext_vector_type(16))) _Float16 v16h;
typedef __attribute__((ext_vector_type(8)))  _Float16 h8v;
typedef __attribute__((ext_vector_type(8)))  float    v8f;
typedef unsigned int v4u __attribute__((ext_vector_type(4)));
typedef int          v8i __attribute__((ext_vector_type(8)));
typedef int          v4i __attribute__((ext_vector_type(4)));

#define Bb  2
#define Cc  64
#define Dd  32
#define Hh  96
#define Ww  96
#define HWc (Hh*Ww)            // 9216
#define DHW (Dd*HWc)           // 294912
#define TOTE ((long)Bb*Cc*DHW) // 37748736 elements per tensor

__device__ __forceinline__ v8f wmma_f16(v16h a, v16h b, v8f c) {
  // D = A(16x32 f16) * B(32x16 f16) + C(16x16 f32)
  return __builtin_amdgcn_wmma_f32_16x16x32_f16(false, a, false, b, (short)0, c, false, false);
}

// A fragment from row-major f16 [M][lda]: A[m][k] contiguous in k.
__device__ __forceinline__ v16h load_a16(const f16* A, int lda, int m0, int k0, int lane) {
  int hf = lane >> 4;
  const f16* row = A + (m0 + (lane & 15)) * lda + k0 + 8 * hf;
  h8v lo = *(const h8v*)(row);        // K = k0+8h .. +7
  h8v hi = *(const h8v*)(row + 16);   // K = k0+16+8h .. +7
  v16h r;
#pragma unroll
  for (int i = 0; i < 8; ++i) { r[i] = lo[i]; r[8 + i] = hi[i]; }
  return r;
}

// B fragment where logical B[k][n] is stored as Bsrc[n][k] row-major (stride ldb).
__device__ __forceinline__ v16h load_bT(const f16* Bsrc, int ldb, int n0, int k0, int lane) {
  const f16* row = Bsrc + (n0 + (lane & 15)) * ldb + k0 + 8 * (lane >> 4);
  h8v lo = *(const h8v*)(row);
  h8v hi = *(const h8v*)(row + 16);
  v16h r;
#pragma unroll
  for (int i = 0; i < 8; ++i) { r[i] = lo[i]; r[8 + i] = hi[i]; }
  return r;
}

// A fragment where logical A[m][k] is stored as f32 X[k][m] (stride ldx); convert on the fly.
__device__ __forceinline__ v16h load_aT_f32(const float* X, int ldx, int m0, int k0, int lane) {
  int hf = lane >> 4, m = m0 + (lane & 15);
  v16h r;
#pragma unroll
  for (int i = 0; i < 8; ++i) {
    r[i]     = (f16)X[(k0 + 8 * hf + i) * ldx + m];
    r[8 + i] = (f16)X[(k0 + 16 + 8 * hf + i) * ldx + m];
  }
  return r;
}

// TDM: DMA one 64x96-float tile (row stride DHW elements) from global into LDS.
// D# built per CDNA5 ISA; 2D tensor -> groups 2/3 zero.
// This toolchain exposes the 6-arg builtin (extra int32x8 group before cpol).
__device__ __forceinline__ void tdm_load_row(unsigned lds_off, const float* gptr) {
  unsigned long long ga = (unsigned long long)(uintptr_t)gptr;
  v4u g0;
  g0[0] = 1u;                                            // count=1 (valid), user mode
  g0[1] = lds_off;                                       // lds_addr (bytes)
  g0[2] = (unsigned)(ga & 0xFFFFFFFFull);                // global_addr[31:0]
  g0[3] = (unsigned)((ga >> 32) & 0x1FFFFFFull) | (2u << 30); // addr[56:32] | type=2
  v8i g1;
  g1[0] = 0x20000;                                       // data_size=2 (4 bytes)
  g1[1] = (int)(96u << 16);                              // tensor_dim0=96 (bits 79:48)
  g1[2] = (int)(64u << 16);                              // tensor_dim1=64 (bits 111:80)
  g1[3] = (int)(96u << 16);                              // tile_dim0=96 (bits 127:112)
  g1[4] = 64;                                            // tile_dim1=64, tile_dim2=0
  g1[5] = DHW;                                           // tensor_dim0_stride lo (elements)
  g1[6] = 0;                                             // stride0 hi | stride1 lo
  g1[7] = 0;                                             // stride1 hi
  v4i z4 = {0, 0, 0, 0};
  v8i z8 = {0, 0, 0, 0, 0, 0, 0, 0};
  __builtin_amdgcn_tensor_load_to_lds(g0, g1, z4, z4, z8, 0);
}

__global__ __launch_bounds__(256)
void scam_row_kernel(const float* __restrict__ x_l, const float* __restrict__ x_r,
                     const float* __restrict__ nlw, const float* __restrict__ nlb,
                     const float* __restrict__ nrw, const float* __restrict__ nrb,
                     const float* __restrict__ l1w, const float* __restrict__ l1b,
                     const float* __restrict__ r1w, const float* __restrict__ r1b,
                     const float* __restrict__ l2w, const float* __restrict__ l2b,
                     const float* __restrict__ r2w, const float* __restrict__ r2b,
                     const float* __restrict__ beta, const float* __restrict__ gamma,
                     float* __restrict__ out)
{
  // ---- LDS regions (reused across phases) ----
  __shared__ __align__(16) float R1[12288];   // xl[c][w],xr[c][w] -> S[96][96] -> F_r2l,F_l2r [c][w]
  __shared__ __align__(16) f16   R2[12288];   // XNl[96][64],XNr[96][64] -> P1[96][96]
  __shared__ __align__(16) f16   R5[16384];   // W1l,W1r,W2l,W2r [64][64]  -> P2t[96][96]
  __shared__ __align__(16) f16   Ql_s[6144];  // [w][c]
  __shared__ __align__(16) f16   Qr_s[6144];  // [v][c]
  __shared__ __align__(16) f16   Vlt_s[6144]; // [c][w]
  __shared__ __align__(16) f16   Vrt_s[6144]; // [c][v]
  __shared__ __align__(16) float small_s[640];

  float* xl_s = R1;            // [64][96]
  float* xr_s = R1 + 6144;
  float* S_s  = R1;            // [96][96] logits (after x dead)
  f16* XNl = R2;               // [96][64]
  f16* XNr = R2 + 6144;
  f16* P1  = R2;               // [96][96] row-softmax probs
  f16* W1l = R5, *W1r = R5 + 4096, *W2l = R5 + 8192, *W2r = R5 + 12288;
  f16* P2t = R5;               // [96][96] col-softmax probs, transposed [v][w]

  const int tid  = threadIdx.x, lane = tid & 31;
  const int wvu  = __builtin_amdgcn_readfirstlane(tid >> 5);  // wave id as SGPR -> scalar branches
  const int bid  = blockIdx.x;
  const int h = bid % Hh, d = (bid / Hh) % Dd, b = bid / (Hh * Dd);
  const long base = (long)b * ((long)Cc * DHW) + (long)d * HWc + (long)h * Ww;

  // ---- Phase 1: TDM DMA for x rows (wave 0), weights/bias via VGPRs (all waves) ----
  if (wvu == 0) {
    tdm_load_row((unsigned)(uintptr_t)xl_s, x_l + base);
    tdm_load_row((unsigned)(uintptr_t)xr_s, x_r + base);
  }
  if (tid < 64) {
    small_s[tid]       = nlw[tid];  small_s[64  + tid] = nlb[tid];
    small_s[128 + tid] = nrw[tid];  small_s[192 + tid] = nrb[tid];
    small_s[256 + tid] = l1b[tid];  small_s[320 + tid] = r1b[tid];
    small_s[384 + tid] = l2b[tid];  small_s[448 + tid] = r2b[tid];
    small_s[512 + tid] = beta[tid]; small_s[576 + tid] = gamma[tid];
  }
  for (int i = tid; i < 4096; i += 256) {
    W1l[i] = (f16)l1w[i]; W1r[i] = (f16)r1w[i];
    W2l[i] = (f16)l2w[i]; W2r[i] = (f16)r2w[i];
  }
  if (wvu == 0) __builtin_amdgcn_s_wait_tensorcnt(0);  // x tiles resident in LDS
  __syncthreads();

  // ---- Phase 2: channel LayerNorm (fp32 stats), write f16 XN[w][c] ----
  if (tid < 192) {
    int side = tid / Ww, w = tid % Ww;
    const float* xs = side ? xr_s : xl_s;
    f16* xn = side ? XNr : XNl;
    const float* nw = small_s + (side ? 128 : 0);
    const float* nb = small_s + (side ? 192 : 64);
    float s = 0.f, s2 = 0.f;
    for (int c = 0; c < Cc; ++c) { float v = xs[c * Ww + w]; s += v; s2 += v * v; }
    float mu = s * (1.f / 64.f);
    float var = s2 * (1.f / 64.f) - mu * mu;
    float rs = rsqrtf(var + 1e-5f);
    for (int c = 0; c < Cc; ++c)
      xn[w * 64 + c] = (f16)(((xs[c * Ww + w] - mu) * rs) * nw[c] + nb[c]);
  }
  __syncthreads();

  // ---- Phase 3: projections Q_l,Q_r (from XN) and V_l,V_r (from raw x), 96x64 @ 64x64 ----
  for (int job = wvu; job < 96; job += 8) {
    int g = job / 24, r = job % 24, m0 = (r >> 2) * 16, n0 = (r & 3) * 16;
    const f16* Wsrc = (g == 0) ? W1l : (g == 1) ? W1r : (g == 2) ? W2l : W2r;
    v8f acc = {};
#pragma unroll
    for (int k0 = 0; k0 < 64; k0 += 32) {
      v16h a;
      if      (g == 0) a = load_a16(XNl, 64, m0, k0, lane);
      else if (g == 1) a = load_a16(XNr, 64, m0, k0, lane);
      else if (g == 2) a = load_aT_f32(xl_s, Ww, m0, k0, lane);
      else             a = load_aT_f32(xr_s, Ww, m0, k0, lane);
      acc = wmma_f16(a, load_bT(Wsrc, 64, n0, k0, lane), acc);
    }
    int hf = lane >> 4, n = n0 + (lane & 15);
    float bn = small_s[256 + g * 64 + n];
    if (g < 2) {
      f16* Qd = g ? Qr_s : Ql_s;
#pragma unroll
      for (int j = 0; j < 8; ++j) Qd[(m0 + j + 8 * hf) * 64 + n] = (f16)(acc[j] + bn);
    } else {
      f16* Vd = (g == 2) ? Vlt_s : Vrt_s;   // store transposed [c][w]
#pragma unroll
      for (int j = 0; j < 8; ++j) Vd[n * Ww + (m0 + j + 8 * hf)] = (f16)(acc[j] + bn);
    }
  }
  __syncthreads();

  // ---- Phase 4: S = (Q_l @ Q_r^T) * C^-0.5  (96x96, K=64) ----
  for (int job = wvu; job < 36; job += 8) {
    int m0 = (job / 6) * 16, n0 = (job % 6) * 16;
    v8f acc = {};
#pragma unroll
    for (int k0 = 0; k0 < 64; k0 += 32)
      acc = wmma_f16(load_a16(Ql_s, 64, m0, k0, lane), load_bT(Qr_s, 64, n0, k0, lane), acc);
    int hf = lane >> 4, n = n0 + (lane & 15);
#pragma unroll
    for (int j = 0; j < 8; ++j) S_s[(m0 + j + 8 * hf) * Ww + n] = acc[j] * 0.125f;
  }
  __syncthreads();

  // ---- Phase 5: row softmax -> P1[w][v]; column softmax -> P2t[v][w] ----
  if (tid < 96) {
    int w = tid;
    float mx = -1e30f;
    for (int v = 0; v < Ww; ++v) mx = fmaxf(mx, S_s[w * Ww + v]);
    float sum = 0.f;
    for (int v = 0; v < Ww; ++v) sum += __expf(S_s[w * Ww + v] - mx);
    float inv = 1.f / sum;
    for (int v = 0; v < Ww; ++v) P1[w * Ww + v] = (f16)(__expf(S_s[w * Ww + v] - mx) * inv);
  } else if (tid < 192) {
    int v = tid - 96;
    float mx = -1e30f;
    for (int w = 0; w < Ww; ++w) mx = fmaxf(mx, S_s[w * Ww + v]);
    float sum = 0.f;
    for (int w = 0; w < Ww; ++w) sum += __expf(S_s[w * Ww + v] - mx);
    float inv = 1.f / sum;
    for (int w = 0; w < Ww; ++w) P2t[v * Ww + w] = (f16)(__expf(S_s[w * Ww + v] - mx) * inv);
  }
  __syncthreads();

  // ---- Phase 6: F_r2l = P1 @ V_r ; F_l2r = P2^T @ V_l  (96x64, K=96) -> F[c][w] f32 in R1 ----
  float* F_r2l = R1;           // [64][96]
  float* F_l2r = R1 + 6144;    // [64][96]
  for (int job = wvu; job < 48; job += 8) {
    int g = job / 24, r = job % 24, m0 = (r >> 2) * 16, n0 = (r & 3) * 16;
    const f16* A  = g ? P2t   : P1;
    const f16* Bv = g ? Vlt_s : Vrt_s;
    float* Fd = g ? F_l2r : F_r2l;
    v8f acc = {};
#pragma unroll
    for (int k0 = 0; k0 < 96; k0 += 32)
      acc = wmma_f16(load_a16(A, Ww, m0, k0, lane), load_bT(Bv, Ww, n0, k0, lane), acc);
    int hf = lane >> 4, n = n0 + (lane & 15);
#pragma unroll
    for (int j = 0; j < 8; ++j) Fd[n * Ww + (m0 + j + 8 * hf)] = acc[j];
  }
  __syncthreads();

  // ---- Phase 7: residual epilogue, coalesced over w (x re-read hits L2) ----
  for (int i = tid; i < Cc * Ww; i += 256) {
    int c = i / Ww, w = i - c * Ww;
    long g = base + (long)c * DHW + w;
    out[g]        = x_l[g] + F_r2l[i] * small_s[512 + c];
    out[TOTE + g] = x_r[g] + F_l2r[i] * small_s[576 + c];
  }
}

extern "C" void kernel_launch(void* const* d_in, const int* in_sizes, int n_in,
                              void* d_out, int out_size, void* d_ws, size_t ws_size,
                              hipStream_t stream) {
  (void)in_sizes; (void)n_in; (void)out_size; (void)d_ws; (void)ws_size;
  dim3 grid(Bb * Dd * Hh);   // 6144 blocks, one per (b,d,h) row
  dim3 block(256);           // 8 wave32 waves
  scam_row_kernel<<<grid, block, 0, stream>>>(
      (const float*)d_in[0],  (const float*)d_in[1],
      (const float*)d_in[2],  (const float*)d_in[3],
      (const float*)d_in[4],  (const float*)d_in[5],
      (const float*)d_in[6],  (const float*)d_in[7],
      (const float*)d_in[8],  (const float*)d_in[9],
      (const float*)d_in[10], (const float*)d_in[11],
      (const float*)d_in[12], (const float*)d_in[13],
      (const float*)d_in[14], (const float*)d_in[15],
      (float*)d_out);
}
